// TransLayer_7834020348080
// MI455X (gfx1250) — compile-verified, matrix operand order checked by reading, not run on hardware
//
#include <hip/hip_runtime.h>

#define DEV __device__ __forceinline__

typedef __attribute__((ext_vector_type(16))) _Float16 v16h;
typedef __attribute__((ext_vector_type(8)))  _Float16 v8h;
typedef __attribute__((ext_vector_type(8)))  float    v8f;

// 16-bit A/B fragment from contiguous-K storage: lane L holds row (L&15),
// K values {kh..kh+7, 16+kh..16+kh+7} with kh = (L>>4)*8. Two 16B loads.
DEV v16h frag_contig(const _Float16* p) {
  v8h lo = *(const v8h*)(p);
  v8h hi = *(const v8h*)(p + 16);
  return __builtin_shufflevector(lo, hi, 0,1,2,3,4,5,6,7,8,9,10,11,12,13,14,15);
}

DEV v8f wmma16(v16h a, v16h b, v8f c) {
  return __builtin_amdgcn_wmma_f32_16x16x32_f16(false, a, false, b, (short)0, c,
                                                false, false);
}

// ---------------------------------------------------------------- LayerNorm
__global__ void ln_kernel(const float* __restrict__ x, const float* __restrict__ g,
                          const float* __restrict__ bta, _Float16* __restrict__ xn) {
  __shared__ float s1[256], s2[256];
  const int row = blockIdx.x, t = threadIdx.x;
  const float* p = x + (size_t)row * 512;
  float a = p[t], b = p[t + 256];
  s1[t] = a + b; s2[t] = a * a + b * b;
  __syncthreads();
  for (int s = 128; s > 0; s >>= 1) {
    if (t < s) { s1[t] += s1[t + s]; s2[t] += s2[t + s]; }
    __syncthreads();
  }
  float mu = s1[0] * (1.f / 512.f);
  float var = s2[0] * (1.f / 512.f) - mu * mu;
  float rstd = rsqrtf(var + 1e-5f);
  _Float16* q = xn + (size_t)row * 512;
  q[t]       = (_Float16)((a - mu) * rstd * g[t]       + bta[t]);
  q[t + 256] = (_Float16)((b - mu) * rstd * g[t + 256] + bta[t + 256]);
}

// ------------------------------------------------- weight convert/transpose
__global__ void prep_w_kernel(const float* __restrict__ wqkv, const float* __restrict__ wout,
                              _Float16* __restrict__ WqkvT, _Float16* __restrict__ WoutT) {
  int i = blockIdx.x * 256 + threadIdx.x;          // covers 512*1536 + 512*512
  if (i < 512 * 1536) {
    int r = i / 1536, c = i % 1536;
    WqkvT[(size_t)c * 512 + r] = (_Float16)wqkv[i];
  } else {
    int j = i - 512 * 1536;
    int r = j / 512, c = j % 512;
    WoutT[(size_t)c * 512 + r] = (_Float16)wout[j];
  }
}

// -------------------------------------------- QKV GEMM -> [b,h,n,64] scatter
__global__ void qkv_gemm_kernel(const _Float16* __restrict__ XN,
                                const _Float16* __restrict__ WqkvT,
                                _Float16* __restrict__ Q, _Float16* __restrict__ K,
                                _Float16* __restrict__ V) {
  const int lane = threadIdx.x & 31, wave = threadIdx.x >> 5;
  const int row0 = blockIdx.x * 128 + wave * 16;
  const int col0 = blockIdx.y * 64;
  const int m = lane & 15, kh = (lane >> 4) * 8;
  v8f acc[4] = {};
  for (int kb = 0; kb < 512; kb += 32) {
    v16h a = frag_contig(XN + (size_t)(row0 + m) * 512 + kb + kh);
#pragma unroll
    for (int s = 0; s < 4; ++s) {
      v16h b = frag_contig(WqkvT + (size_t)(col0 + s * 16 + m) * 512 + kb + kh);
      acc[s] = wmma16(a, b, acc[s]);
    }
  }
  const int rbase = row0 + 8 * (lane >> 4);
#pragma unroll
  for (int s = 0; s < 4; ++s) {
    int col = col0 + s * 16 + (lane & 15);
    int sect = col >> 9, inner = col & 511;
    int h = inner >> 6, d = inner & 63;
    _Float16* dst = sect == 0 ? Q : (sect == 1 ? K : V);
    float scale = sect == 0 ? 0.125f : 1.f;   // DH^-0.5 folded into q
#pragma unroll
    for (int r = 0; r < 8; ++r) {
      int row = rbase + r;
      int bb = row >> 12, nn = row & 4095;
      dst[(((size_t)(bb * 8 + h) * 4096) + nn) * 64 + d] = (_Float16)(acc[s][r] * scale);
    }
  }
}

// --------------------------------------------------------------- landmarks
__global__ void landmark_kernel(const _Float16* __restrict__ Q, const _Float16* __restrict__ K,
                                _Float16* __restrict__ QL, _Float16* __restrict__ KL) {
  int i = blockIdx.x * 256 + threadIdx.x;          // 32*256*64 threads
  int d = i & 63, mm = (i >> 6) & 255, bh = i >> 14;
  size_t src = ((size_t)bh * 4096 + mm * 16) * 64 + d;
  float sq = 0.f, sk = 0.f;
#pragma unroll
  for (int t = 0; t < 16; ++t) {
    sq += (float)Q[src + (size_t)t * 64];
    sk += (float)K[src + (size_t)t * 64];
  }
  size_t dst = ((size_t)bh * 256 + mm) * 64 + d;
  QL[dst] = (_Float16)(sq * (1.f / 16.f));
  KL[dst] = (_Float16)(sk * (1.f / 16.f));
}

// ---------------------------- batched GEMM, B transposed (Bt row-major [N,K])
template <int KD>
__global__ void gemm_nt_kernel(const _Float16* __restrict__ A, int lda, long long sA,
                               const _Float16* __restrict__ Bt, int ldb, long long sB,
                               _Float16* __restrict__ C, int ldc, long long sC) {
  A  += (size_t)blockIdx.z * sA;
  Bt += (size_t)blockIdx.z * sB;
  C  += (size_t)blockIdx.z * sC;
  const int lane = threadIdx.x & 31, wave = threadIdx.x >> 5;
  const int row0 = blockIdx.x * 128 + wave * 16;
  const int col0 = blockIdx.y * 64;
  const int m = lane & 15, kh = (lane >> 4) * 8;
  v8f acc[4] = {};
#pragma unroll
  for (int kb = 0; kb < KD; kb += 32) {
    v16h a = frag_contig(A + (size_t)(row0 + m) * lda + kb + kh);
#pragma unroll
    for (int s = 0; s < 4; ++s) {
      v16h b = frag_contig(Bt + (size_t)(col0 + s * 16 + m) * ldb + kb + kh);
      acc[s] = wmma16(a, b, acc[s]);
    }
  }
  const int rbase = row0 + 8 * (lane >> 4);
#pragma unroll
  for (int s = 0; s < 4; ++s) {
    int col = col0 + s * 16 + (lane & 15);
#pragma unroll
    for (int r = 0; r < 8; ++r)
      C[(size_t)(rbase + r) * ldc + col] = (_Float16)acc[s][r];
  }
}

// ------------------- batched GEMM, B row-major [K,N]; postops for pinv chain
// B k-slab (32 x 64) staged into LDS transposed so WMMA fragments read as two
// contiguous 16B ds_load_b128 per fragment (row stride 40 halves kills the
// 16-lane bank-conflict pattern). Coalesced 16B global loads for staging.
// postop 0: C = scale*acc    1: C = alpha*I - acc    3: merged-head store
template <int KD>
__global__ void gemm_nn_kernel(const _Float16* __restrict__ A, int lda, long long sA,
                               const _Float16* __restrict__ B, int ldb, long long sB,
                               _Float16* __restrict__ C, int ldc, long long sC,
                               int postop, float alpha, float scale) {
  __shared__ _Float16 BsT[64 * 40];                 // [n][k] transposed, padded
  const int batch = blockIdx.z;
  A += (size_t)batch * sA;
  B += (size_t)batch * sB;
  if (postop == 3) {  // C laid out [b, n, 512]; batch bh writes columns h*64..h*64+63
    C += ((size_t)(batch >> 3) * 4096) * 512 + (size_t)(batch & 7) * 64;
    ldc = 512;
  } else {
    C += (size_t)batch * sC;
  }
  const int lane = threadIdx.x & 31, wave = threadIdx.x >> 5;
  const int row0 = blockIdx.x * 128 + wave * 16;
  const int col0 = blockIdx.y * 64;
  const int m = lane & 15, kh = (lane >> 4) * 8;
  const int krow = threadIdx.x >> 3;                // 0..31: K row within slab
  const int ncol = (threadIdx.x & 7) * 8;           // 0..56: N col within slab
  v8f acc[4] = {};
  for (int kb = 0; kb < KD; kb += 32) {
    v8h bv = *(const v8h*)(B + (size_t)(kb + krow) * ldb + col0 + ncol);
    __syncthreads();                                // previous slab fully read
#pragma unroll
    for (int i = 0; i < 8; ++i)
      BsT[(ncol + i) * 40 + krow] = bv[i];
    __syncthreads();
    v16h a = frag_contig(A + (size_t)(row0 + m) * lda + kb + kh);
#pragma unroll
    for (int s = 0; s < 4; ++s) {
      const _Float16* bp = &BsT[(s * 16 + m) * 40 + kh];
      v8h lo = *(const v8h*)(bp);
      v8h hi = *(const v8h*)(bp + 16);
      v16h b = __builtin_shufflevector(lo, hi, 0,1,2,3,4,5,6,7,8,9,10,11,12,13,14,15);
      acc[s] = wmma16(a, b, acc[s]);
    }
  }
  const int rbase = row0 + 8 * (lane >> 4);
#pragma unroll
  for (int s = 0; s < 4; ++s) {
    int col = col0 + s * 16 + (lane & 15);
#pragma unroll
    for (int r = 0; r < 8; ++r) {
      int row = rbase + r;
      float v = acc[s][r];
      float o = (postop == 1) ? (alpha * (row == col ? 1.f : 0.f) - v) : v * scale;
      C[(size_t)row * ldc + col] = (_Float16)o;
    }
  }
}

// ------------------------------------------------------- row softmax, f16 io
__global__ void softmax_rows_kernel(_Float16* __restrict__ X, int cols) {
  __shared__ float red[256];
  _Float16* p = X + (size_t)blockIdx.x * cols;
  const int t = threadIdx.x;
  float mx = -3.0e38f;
  for (int j = t; j < cols; j += 256) mx = fmaxf(mx, (float)p[j]);
  red[t] = mx; __syncthreads();
  for (int s = 128; s > 0; s >>= 1) { if (t < s) red[t] = fmaxf(red[t], red[t + s]); __syncthreads(); }
  mx = red[0]; __syncthreads();
  float sum = 0.f;
  for (int j = t; j < cols; j += 256) sum += __expf((float)p[j] - mx);
  red[t] = sum; __syncthreads();
  for (int s = 128; s > 0; s >>= 1) { if (t < s) red[t] += red[t + s]; __syncthreads(); }
  float inv = 1.f / red[0];
  for (int j = t; j < cols; j += 256) p[j] = (_Float16)(__expf((float)p[j] - mx) * inv);
}

// --------------- pinv init: denom = max colsum * max rowsum; Z0 = A2^T/denom
__global__ void pinv_init_kernel(const _Float16* __restrict__ A2, _Float16* __restrict__ Z) {
  __shared__ float rc[256], rr[256];
  const int bh = blockIdx.x, t = threadIdx.x;
  const _Float16* X = A2 + (size_t)bh * 65536;
  _Float16* Zo = Z + (size_t)bh * 65536;
  float cs = 0.f, rs = 0.f;
  for (int i = 0; i < 256; ++i) {
    cs += (float)X[(size_t)i * 256 + t];
    rs += (float)X[(size_t)t * 256 + i];
  }
  rc[t] = cs; rr[t] = rs; __syncthreads();
  for (int s = 128; s > 0; s >>= 1) {
    if (t < s) { rc[t] = fmaxf(rc[t], rc[t + s]); rr[t] = fmaxf(rr[t], rr[t + s]); }
    __syncthreads();
  }
  float inv = 1.f / (rc[0] * rr[0]);
  for (int i = 0; i < 256; ++i)
    Zo[(size_t)t * 256 + i] = (_Float16)((float)X[(size_t)i * 256 + t] * inv);
}

// ------------------------------------------------------ Out = alpha*I - In
__global__ void alphaI_minus_kernel(const _Float16* __restrict__ In,
                                    _Float16* __restrict__ Out, float alpha) {
  size_t i = (size_t)blockIdx.x * 256 + threadIdx.x;   // 32*256*256
  int r = (int)((i >> 8) & 255), c = (int)(i & 255);
  Out[i] = (_Float16)(alpha * (r == c ? 1.f : 0.f) - (float)In[i]);
}

// ------------------------- depthwise conv over n (K=33, pad 16) added to OUTH
__global__ void conv_add_kernel(const _Float16* __restrict__ V, const float* __restrict__ rw,
                                _Float16* __restrict__ OUTH) {
  int i = blockIdx.x * 256 + threadIdx.x;              // 32*4096*64
  int d = i & 63, nn = (i >> 6) & 4095, bh = i >> 18;
  int b = bh >> 3, h = bh & 7;
  const _Float16* v = V + ((size_t)bh * 4096) * 64 + d;
  float acc = 0.f;
#pragma unroll
  for (int t = 0; t < 33; ++t) {
    int s = nn + t - 16;
    if (s >= 0 && s < 4096) acc += rw[h * 33 + t] * (float)v[(size_t)s * 64];
  }
  size_t o = ((size_t)b * 4096 + nn) * 512 + (size_t)h * 64 + d;
  OUTH[o] = (_Float16)((float)OUTH[o] + acc);
}

// ----------------------- final projection: out = x + OUTH @ Wout^T + b_out
__global__ void proj_gemm_kernel(const _Float16* __restrict__ OUTH,
                                 const _Float16* __restrict__ WoutT,
                                 const float* __restrict__ X, const float* __restrict__ bout,
                                 float* __restrict__ Out) {
  const int lane = threadIdx.x & 31, wave = threadIdx.x >> 5;
  const int row0 = blockIdx.x * 128 + wave * 16;
  const int col0 = blockIdx.y * 64;
  const int m = lane & 15, kh = (lane >> 4) * 8;
  v8f acc[4] = {};
  for (int kb = 0; kb < 512; kb += 32) {
    v16h a = frag_contig(OUTH + (size_t)(row0 + m) * 512 + kb + kh);
#pragma unroll
    for (int s = 0; s < 4; ++s) {
      v16h b = frag_contig(WoutT + (size_t)(col0 + s * 16 + m) * 512 + kb + kh);
      acc[s] = wmma16(a, b, acc[s]);
    }
  }
  const int rbase = row0 + 8 * (lane >> 4);
#pragma unroll
  for (int s = 0; s < 4; ++s) {
    int col = col0 + s * 16 + (lane & 15);
    float bo = bout[col];
#pragma unroll
    for (int r = 0; r < 8; ++r) {
      size_t idx = (size_t)(rbase + r) * 512 + col;
      Out[idx] = acc[s][r] + bo + X[idx];
    }
  }
}

extern "C" void kernel_launch(void* const* d_in, const int* in_sizes, int n_in,
                              void* d_out, int out_size, void* d_ws, size_t ws_size,
                              hipStream_t stream) {
  (void)in_sizes; (void)n_in; (void)out_size; (void)ws_size;
  const float* x    = (const float*)d_in[0];
  const float* ln_g = (const float*)d_in[1];
  const float* ln_b = (const float*)d_in[2];
  const float* wqkv = (const float*)d_in[3];
  const float* wout = (const float*)d_in[4];
  const float* bout = (const float*)d_in[5];
  const float* resw = (const float*)d_in[6];
  float* out = (float*)d_out;

  char* base = (char*)d_ws;
  size_t off = 0;
  auto alloc = [&](size_t bytes) -> void* {
    void* p = base + off;
    off = (off + bytes + 255) & ~(size_t)255;
    return p;
  };

  _Float16* XN    = (_Float16*)alloc(16384ull * 512 * 2);    // aliased as OUTH later
  _Float16* Q     = (_Float16*)alloc(32ull * 4096 * 64 * 2);
  _Float16* K     = (_Float16*)alloc(32ull * 4096 * 64 * 2);
  _Float16* V     = (_Float16*)alloc(32ull * 4096 * 64 * 2);
  _Float16* QL    = (_Float16*)alloc(32ull * 256 * 64 * 2);
  _Float16* KL    = (_Float16*)alloc(32ull * 256 * 64 * 2);
  _Float16* WqkvT = (_Float16*)alloc(1536ull * 512 * 2);
  _Float16* WoutT = (_Float16*)alloc(512ull * 512 * 2);
  _Float16* A1    = (_Float16*)alloc(32ull * 4096 * 256 * 2);
  _Float16* A2    = (_Float16*)alloc(32ull * 256 * 256 * 2);
  _Float16* Zb    = (_Float16*)alloc(32ull * 256 * 256 * 2);
  _Float16* Z2    = (_Float16*)alloc(32ull * 256 * 256 * 2);
  _Float16* XZ    = (_Float16*)alloc(32ull * 256 * 256 * 2);
  _Float16* U     = (_Float16*)alloc(32ull * 256 * 256 * 2);  // also holds P
  _Float16* W     = (_Float16*)alloc(32ull * 256 * 256 * 2);
  _Float16* A3    = (_Float16*)alloc(32ull * 256 * 4096 * 2); // reused for A1@pinv
  _Float16* A3V   = (_Float16*)alloc(32ull * 256 * 64 * 2);
  _Float16* OUTH  = XN;                                       // XN dead after qkv GEMM

  prep_w_kernel<<<4096, 256, 0, stream>>>(wqkv, wout, WqkvT, WoutT);
  ln_kernel<<<16384, 256, 0, stream>>>(x, ln_g, ln_b, XN);
  qkv_gemm_kernel<<<dim3(128, 24), 256, 0, stream>>>(XN, WqkvT, Q, K, V);
  landmark_kernel<<<2048, 256, 0, stream>>>(Q, K, QL, KL);

  // attn1 = softmax(Q @ KL^T)  [4096,256] per bh
  gemm_nt_kernel<64><<<dim3(32, 4, 32), 256, 0, stream>>>(Q, 64, 4096ll * 64, KL, 64, 256ll * 64,
                                                          A1, 256, 4096ll * 256);
  softmax_rows_kernel<<<32 * 4096, 256, 0, stream>>>(A1, 256);

  // attn2 = softmax(QL @ KL^T)  [256,256]
  gemm_nt_kernel<64><<<dim3(2, 4, 32), 256, 0, stream>>>(QL, 64, 256ll * 64, KL, 64, 256ll * 64,
                                                         A2, 256, 256ll * 256);
  softmax_rows_kernel<<<32 * 256, 256, 0, stream>>>(A2, 256);
  pinv_init_kernel<<<32, 256, 0, stream>>>(A2, Zb);

  // Moore-Penrose iterations: z = 0.25 z (13I - xz(15I - xz(7I - xz)))
  _Float16* Zc = Zb; _Float16* Zn = Z2;
  for (int it = 0; it < 6; ++it) {
    gemm_nn_kernel<256><<<dim3(2, 4, 32), 256, 0, stream>>>(A2, 256, 65536ll, Zc, 256, 65536ll,
                                                            XZ, 256, 65536ll, 0, 0.f, 1.f);
    alphaI_minus_kernel<<<8192, 256, 0, stream>>>(XZ, U, 7.f);
    gemm_nn_kernel<256><<<dim3(2, 4, 32), 256, 0, stream>>>(XZ, 256, 65536ll, U, 256, 65536ll,
                                                            W, 256, 65536ll, 1, 15.f, 1.f);
    gemm_nn_kernel<256><<<dim3(2, 4, 32), 256, 0, stream>>>(XZ, 256, 65536ll, W, 256, 65536ll,
                                                            U, 256, 65536ll, 1, 13.f, 1.f);
    gemm_nn_kernel<256><<<dim3(2, 4, 32), 256, 0, stream>>>(Zc, 256, 65536ll, U, 256, 65536ll,
                                                            Zn, 256, 65536ll, 0, 0.f, 0.25f);
    _Float16* tmp = Zc; Zc = Zn; Zn = tmp;
  }

  // attn3 = softmax(QL @ K^T)  [256,4096]
  gemm_nt_kernel<64><<<dim3(2, 64, 32), 256, 0, stream>>>(QL, 64, 256ll * 64, K, 64, 4096ll * 64,
                                                          A3, 4096, 256ll * 4096);
  softmax_rows_kernel<<<32 * 256, 256, 0, stream>>>(A3, 4096);

  // A3V = attn3 @ V   [256,64]
  gemm_nn_kernel<4096><<<dim3(2, 1, 32), 256, 0, stream>>>(A3, 4096, 256ll * 4096, V, 64, 4096ll * 64,
                                                           A3V, 64, 256ll * 64, 0, 0.f, 1.f);
  // A1Z = attn1 @ pinv  [4096,256]   (into A3 buffer, now free)
  gemm_nn_kernel<256><<<dim3(32, 4, 32), 256, 0, stream>>>(A1, 256, 4096ll * 256, Zc, 256, 65536ll,
                                                           A3, 256, 4096ll * 256, 0, 0.f, 1.f);
  // O = A1Z @ A3V merged into OUTH [b, n, 512]
  gemm_nn_kernel<256><<<dim3(32, 1, 32), 256, 0, stream>>>(A3, 256, 4096ll * 256, A3V, 64, 256ll * 64,
                                                           OUTH, 0, 0, 3, 0.f, 1.f);
  conv_add_kernel<<<32768, 256, 0, stream>>>(V, resw, OUTH);
  proj_gemm_kernel<<<dim3(128, 8), 256, 0, stream>>>(OUTH, WoutT, x, bout, out);
}
